// OptLayer_44581760533210
// MI455X (gfx1250) — matrix operand order repaired and checked by reading, not compiled, not run on hardware
//
#include <hip/hip_runtime.h>

// ---------------------------------------------------------------------------
// Types for CDNA5 WMMA (wave32): v_wmma_f32_16x16x32_f16
// ---------------------------------------------------------------------------
typedef __attribute__((ext_vector_type(16))) _Float16 v16h;
typedef __attribute__((ext_vector_type(8)))  _Float16 v8h;
typedef __attribute__((ext_vector_type(4)))  _Float16 v4h;
typedef __attribute__((ext_vector_type(8)))  float    v8f;

#define B_DIM   8192
#define K_DIM   1024
#define N_DIM   512
#define TOTAL_F 90.0f
#define N_BISECT 60

// ---------------------------------------------------------------------------
// Kernel 1: f32 -> f16 conversion (x and W copies into workspace).
// 4 elements per thread, 16B load / 8B store, fully coalesced.
// ---------------------------------------------------------------------------
__global__ void __launch_bounds__(256) cvt_f32_to_f16_x4(
    const float* __restrict__ src, _Float16* __restrict__ dst, int n4)
{
    int i = blockIdx.x * blockDim.x + threadIdx.x;
    if (i < n4) {
        const float4 v = ((const float4*)src)[i];
        v4h h;
        h[0] = (_Float16)v.x;
        h[1] = (_Float16)v.y;
        h[2] = (_Float16)v.z;
        h[3] = (_Float16)v.w;
        ((v4h*)dst)[i] = h;
    }
}

// ---------------------------------------------------------------------------
// WMMA fragment gathers, per the CDNA5 ISA VGPR layouts (CDNA5 ISA §7.12.2).
//
// A (16x32 f16, row M = lane%16):
//   lanes 0-15 hold K = {0..7, 16..23}, lanes 16-31 hold K = {8..15, 24..31}
//   -> two 16B loads at (k0 + g*8) and (k0 + g*8 + 16), g = lane>>4.
// B (32x16 f16, col N = lane%16, column-major over K = contiguous W row):
//   lanes 0-15 hold K = {0..7, 16..23}, lanes 16-31 hold K = {8..15, 24..31}
//   -> two 16B loads at (k0 + g*8) and (+16).
// ---------------------------------------------------------------------------
__device__ __forceinline__ v16h load_fragA(const _Float16* p)
{
    v8h lo = *(const v8h*)(p);
    v8h hi = *(const v8h*)(p + 16);
    return __builtin_shufflevector(lo, hi, 0,1,2,3,4,5,6,7,8,9,10,11,12,13,14,15);
}

__device__ __forceinline__ v16h load_fragB(const _Float16* p)
{
    v8h lo = *(const v8h*)(p);
    v8h hi = *(const v8h*)(p + 16);
    return __builtin_shufflevector(lo, hi, 0,1,2,3,4,5,6,7,8,9,10,11,12,13,14,15);
}

__device__ __forceinline__ v8f wmma_f16(v16h a, v16h b, v8f c)
{
    // (neg_a, A, neg_b, B, c_mod, C, reuse_a, reuse_b)
    return __builtin_amdgcn_wmma_f32_16x16x32_f16(false, a, false, b,
                                                  (short)0, c, false, false);
}

// ---------------------------------------------------------------------------
// Kernel 2: z = x @ W^T - b  via WMMA.
// Each wave computes a 64(M) x 32(N) macro-tile = 4x2 WMMA tiles.
// Grid covers the output exactly -> EXEC is all-1s (WMMA requirement).
// ---------------------------------------------------------------------------
__global__ void __launch_bounds__(256) gemm_wmma_f16(
    const _Float16* __restrict__ xh,   // [B_DIM, K_DIM] f16
    const _Float16* __restrict__ wh,   // [N_DIM, K_DIM] f16 (row o = column o of B)
    const float*    __restrict__ bias, // [N_DIM]
    float*          __restrict__ z)    // [B_DIM, N_DIM]
{
    const int lane    = threadIdx.x & 31;
    const int wglobal = (blockIdx.x * 256 + threadIdx.x) >> 5;
    const int tn = wglobal & 15;   // 16 N macro-tiles of 32
    const int tm = wglobal >> 4;   // 128 M macro-tiles of 64
    const int m0 = tm * 64;
    const int n0 = tn * 32;
    const int g  = lane >> 4;      // lane half-group
    const int r  = lane & 15;

    const _Float16* a0 = xh + (size_t)(m0 + r) * K_DIM + g * 8;
    const _Float16* a1 = a0 + (size_t)16 * K_DIM;
    const _Float16* a2 = a0 + (size_t)32 * K_DIM;
    const _Float16* a3 = a0 + (size_t)48 * K_DIM;
    const _Float16* b0 = wh + (size_t)(n0 + r) * K_DIM + g * 8;
    const _Float16* b1 = b0 + (size_t)16 * K_DIM;

    v8f acc[4][2];
    #pragma unroll
    for (int i = 0; i < 4; ++i)
        #pragma unroll
        for (int j = 0; j < 2; ++j)
            acc[i][j] = (v8f){0.f,0.f,0.f,0.f,0.f,0.f,0.f,0.f};

    #pragma unroll 4
    for (int k0 = 0; k0 < K_DIM; k0 += 32) {
        v16h af0 = load_fragA(a0 + k0);
        v16h af1 = load_fragA(a1 + k0);
        v16h af2 = load_fragA(a2 + k0);
        v16h af3 = load_fragA(a3 + k0);
        v16h bf0 = load_fragB(b0 + k0);
        v16h bf1 = load_fragB(b1 + k0);

        acc[0][0] = wmma_f16(af0, bf0, acc[0][0]);
        acc[0][1] = wmma_f16(af0, bf1, acc[0][1]);
        acc[1][0] = wmma_f16(af1, bf0, acc[1][0]);
        acc[1][1] = wmma_f16(af1, bf1, acc[1][1]);
        acc[2][0] = wmma_f16(af2, bf0, acc[2][0]);
        acc[2][1] = wmma_f16(af2, bf1, acc[2][1]);
        acc[3][0] = wmma_f16(af3, bf0, acc[3][0]);
        acc[3][1] = wmma_f16(af3, bf1, acc[3][1]);
    }

    // C/D layout: lane L, element e -> row M = e + 8*(L>>4), col N = L&15.
    #pragma unroll
    for (int j = 0; j < 2; ++j) {
        const int   n  = n0 + j * 16 + r;
        const float bj = bias[n];
        #pragma unroll
        for (int i = 0; i < 4; ++i) {
            const int mb = m0 + i * 16 + 8 * g;
            float* zp = z + (size_t)mb * N_DIM + n;
            #pragma unroll
            for (int e = 0; e < 8; ++e)
                zp[(size_t)e * N_DIM] = acc[i][j][e] - bj;
        }
    }
}

// ---------------------------------------------------------------------------
// Kernel 3: in-place capped-simplex projection, one wave32 per row.
// Each lane holds 16 of the 512 row elements; reductions via __shfl_xor.
// ---------------------------------------------------------------------------
__device__ __forceinline__ float wave_sum(float v)
{
    #pragma unroll
    for (int m = 16; m >= 1; m >>= 1) v += __shfl_xor(v, m, 32);
    return v;
}
__device__ __forceinline__ float wave_min(float v)
{
    #pragma unroll
    for (int m = 16; m >= 1; m >>= 1) v = fminf(v, __shfl_xor(v, m, 32));
    return v;
}
__device__ __forceinline__ float wave_max(float v)
{
    #pragma unroll
    for (int m = 16; m >= 1; m >>= 1) v = fmaxf(v, __shfl_xor(v, m, 32));
    return v;
}

__global__ void __launch_bounds__(256) project_rows(
    float* __restrict__ zy,           // [B_DIM, N_DIM], z in / y out
    const float* __restrict__ u)      // [N_DIM]
{
    const int lane = threadIdx.x & 31;
    const int wave = threadIdx.x >> 5;
    const int row  = blockIdx.x * 8 + wave;
    float* zr = zy + (size_t)row * N_DIM;

    float zv[16], uv[16];
    #pragma unroll
    for (int j = 0; j < 16; ++j) {
        const int idx = lane + 32 * j;
        zv[j] = zr[idx];
        uv[j] = u[idx];
    }

    float zmin = zv[0], zmax = zv[0], umax = uv[0];
    #pragma unroll
    for (int j = 1; j < 16; ++j) {
        zmin = fminf(zmin, zv[j]);
        zmax = fmaxf(zmax, zv[j]);
        umax = fmaxf(umax, uv[j]);
    }
    zmin = wave_min(zmin);
    zmax = wave_max(zmax);
    umax = wave_max(umax);

    float lo = zmin - umax;   // f(lo) = sum(u) - TOTAL >= 0
    float hi = zmax;          // f(hi) = -TOTAL < 0

    for (int it = 0; it < N_BISECT; ++it) {
        const float mid = 0.5f * (lo + hi);
        float s = 0.f;
        #pragma unroll
        for (int j = 0; j < 16; ++j)
            s += fminf(fmaxf(zv[j] - mid, 0.f), uv[j]);
        s = wave_sum(s);
        const float f = s - TOTAL_F;
        if (f > 0.f) lo = mid; else hi = mid;
    }
    const float tau = 0.5f * (lo + hi);

    // Newton correction: tau* = tau + f(tau)/#active  (matches reference)
    float fl = 0.f, kl = 0.f;
    #pragma unroll
    for (int j = 0; j < 16; ++j) {
        const float s = zv[j] - tau;
        fl += fminf(fmaxf(s, 0.f), uv[j]);
        kl += ((s > 0.f) && (s < uv[j])) ? 1.f : 0.f;
    }
    fl = wave_sum(fl) - TOTAL_F;
    kl = fmaxf(wave_sum(kl), 1.f);
    const float tau_star = tau + fl / kl;

    #pragma unroll
    for (int j = 0; j < 16; ++j) {
        const int idx = lane + 32 * j;
        zr[idx] = fminf(fmaxf(zv[idx / 32] - tau_star, 0.f), uv[idx / 32]);
    }
}

// ---------------------------------------------------------------------------
// Host-side launch. Inputs: x[8192,1024] f32, W[512,1024] f32, b[512], u[512].
// Output: y[8192,512] f32. Workspace: xh (16.8 MB) + wh (1 MB) f16 copies.
// ---------------------------------------------------------------------------
extern "C" void kernel_launch(void* const* d_in, const int* in_sizes, int n_in,
                              void* d_out, int out_size, void* d_ws, size_t ws_size,
                              hipStream_t stream)
{
    const float* x = (const float*)d_in[0];
    const float* W = (const float*)d_in[1];
    const float* b = (const float*)d_in[2];
    const float* u = (const float*)d_in[3];
    float* out = (float*)d_out;

    _Float16* xh = (_Float16*)d_ws;
    _Float16* wh = xh + (size_t)B_DIM * K_DIM;

    // 1) f32 -> f16 copies of x and W
    {
        const int n4 = (B_DIM * K_DIM) / 4;
        cvt_f32_to_f16_x4<<<(n4 + 255) / 256, 256, 0, stream>>>(x, xh, n4);
    }
    {
        const int n4 = (N_DIM * K_DIM) / 4;
        cvt_f32_to_f16_x4<<<(n4 + 255) / 256, 256, 0, stream>>>(W, wh, n4);
    }

    // 2) z = x @ W^T - b  (WMMA). 2048 waves, 8 per block -> 256 blocks.
    gemm_wmma_f16<<<256, 256, 0, stream>>>(xh, wh, b, out);

    // 3) capped-simplex projection, one wave per row, in place.
    project_rows<<<B_DIM / 8, 256, 0, stream>>>(out, u);
}
